// QuantLinearTensorWise_17377437680104
// MI455X (gfx1250) — matrix operand order repaired and checked by reading, not compiled, not run on hardware
//
#include <hip/hip_runtime.h>
#include <hip/hip_bf16.h>
#include <stdint.h>

// y[m,n] = scale * sum_k x[m,k] * W[n,k] + bias[n]
// M = B*S = 8192, K = IN_F = 4096, N = OUT_F = 16384
// Memory-bound (~0.94 GB @ 23.3 TB/s ~ 40us vs <1us of bf16 WMMA math).
// bf16 WMMA: W ints in [-127,127] are exact in bf16; scale applied in f32 epilogue.

#define IN_F   4096
#define OUT_F  16384
#define MTOT   8192

#define BM 128
#define BN 128
#define BK 32
#define KTILES (IN_F / BK)   // 128 k-steps

typedef __attribute__((ext_vector_type(16))) __bf16 v16bf;
typedef __attribute__((ext_vector_type(8)))  float  v8f;
typedef int v4i __attribute__((vector_size(16)));   // matches builtin's V4i param

#if defined(__has_builtin)
#  if __has_builtin(__builtin_amdgcn_global_load_async_to_lds_b128)
#    define HAVE_ASYNC_LDS 1
#  else
#    define HAVE_ASYNC_LDS 0
#  endif
#  if __has_builtin(__builtin_amdgcn_s_wait_asynccnt)
#    define HAVE_WAIT_ASYNC 1
#  else
#    define HAVE_WAIT_ASYNC 0
#  endif
#else
#  define HAVE_ASYNC_LDS 0
#  define HAVE_WAIT_ASYNC 0
#endif

// 16-byte global -> LDS copy. Async (ASYNCcnt-tracked direct-to-LDS DMA path)
// when available; otherwise stage through VGPRs (global_load_b128 + ds_store_b128).
__device__ __forceinline__ void copy16_g2l(const void* g, void* l) {
#if HAVE_ASYNC_LDS
  __builtin_amdgcn_global_load_async_to_lds_b128(
      (__attribute__((address_space(1))) v4i*)(void*)g,
      (__attribute__((address_space(3))) v4i*)l,
      /*offset=*/0, /*cpol=*/0);
#else
  *(float4*)l = *(const float4*)g;
#endif
}

__device__ __forceinline__ void wait_async_le8() {
#if HAVE_ASYNC_LDS
#  if HAVE_WAIT_ASYNC
  __builtin_amdgcn_s_wait_asynccnt(8);
#  else
  asm volatile("s_wait_asynccnt 8" ::: "memory");
#  endif
#endif
}

__device__ __forceinline__ void wait_async_0() {
#if HAVE_ASYNC_LDS
#  if HAVE_WAIT_ASYNC
  __builtin_amdgcn_s_wait_asynccnt(0);
#  else
  asm volatile("s_wait_asynccnt 0" ::: "memory");
#  endif
#endif
}

__device__ __forceinline__ void put4f(v16bf& v, int p, float4 f) {
  v[p + 0] = (__bf16)f.x; v[p + 1] = (__bf16)f.y;
  v[p + 2] = (__bf16)f.z; v[p + 3] = (__bf16)f.w;
}
__device__ __forceinline__ void put4i(v16bf& v, int p, int4 f) {
  v[p + 0] = (__bf16)(float)f.x; v[p + 1] = (__bf16)(float)f.y;
  v[p + 2] = (__bf16)(float)f.z; v[p + 3] = (__bf16)(float)f.w;
}

__global__ __launch_bounds__(256)
void qlinear_bf16_wmma(const float* __restrict__ x,
                       const int*   __restrict__ w,
                       const float* __restrict__ scale_p,
                       const float* __restrict__ bias,
                       float*       __restrict__ out) {
  // Double-buffered LDS: x tile kept as f32, W tile as int32 (exact);
  // converted to bf16 while building WMMA fragments. 64 KB total.
  __shared__ float lA[2][BM * BK];
  __shared__ int   lB[2][BN * BK];

  const int n0   = blockIdx.x * BN;
  const int m0   = blockIdx.y * BM;
  const int t    = threadIdx.x;       // 0..255, 8 waves
  const int lane = t & 31;
  const int wave = t >> 5;
  const int half = lane >> 4;         // lane 0-15 vs 16-31 (K-half owner)
  const int l16  = lane & 15;
  const int waveM = (wave & 1) * 64;  // 2 wave-rows of 64 M
  const int waveN = (wave >> 1) * 32; // 4 wave-cols of 32 N

  // Stage one 128x32 x-tile and one 128x32 W-tile into LDS buffer s.
  // 256 threads * 4 chunks * 16B = 16 KB per tile; every access is a
  // coalesced b128 on a fresh cacheline stream (pure-bandwidth path).
  auto stage_in = [&](int s, int kt) {
    const int k0 = kt * BK;
#pragma unroll
    for (int i = 0; i < 4; ++i) {
      const int idx = t + i * 256;     // 0..1023
      const int row = idx >> 3;        // 0..127
      const int q   = (idx & 7) << 2;  // 0,4,...,28
      copy16_g2l(x + (size_t)(m0 + row) * IN_F + k0 + q, &lA[s][row * BK + q]);
    }
#pragma unroll
    for (int i = 0; i < 4; ++i) {
      const int idx = t + i * 256;
      const int row = idx >> 3;
      const int q   = (idx & 7) << 2;
      copy16_g2l(w + (size_t)(n0 + row) * IN_F + k0 + q, &lB[s][row * BK + q]);
    }
  };

  v8f acc[4][2];
#pragma unroll
  for (int i = 0; i < 4; ++i)
#pragma unroll
    for (int j = 0; j < 2; ++j)
      acc[i][j] = (v8f){0.f, 0.f, 0.f, 0.f, 0.f, 0.f, 0.f, 0.f};

  stage_in(0, 0);

  for (int kt = 0; kt < KTILES; ++kt) {
    const int cur = kt & 1;
    if (kt + 1 < KTILES) {
      stage_in(cur ^ 1, kt + 1);  // prefetch next stage (8 async ops/wave)
      wait_async_le8();           // older stage (current) done, newer in flight
    } else {
      wait_async_0();
    }
    __syncthreads();              // current-stage LDS visible to all waves

    // ---- Build bf16 fragments per ISA 7.12.2 layouts ----
    // A 16x32: lane l<16 holds row l, K 0..7 & 16..23; lane>=16 row l-16,
    // K 8..15 & 24..31  -> two 8-float contiguous LDS chunks per lane.
    v16bf afrag[4];
#pragma unroll
    for (int i = 0; i < 4; ++i) {
      const float* ar = &lA[cur][(waveM + i * 16 + l16) * BK + half * 8];
      put4f(afrag[i], 0,  *(const float4*)(ar + 0));
      put4f(afrag[i], 4,  *(const float4*)(ar + 4));
      put4f(afrag[i], 8,  *(const float4*)(ar + 16));
      put4f(afrag[i], 12, *(const float4*)(ar + 20));
    }
    // B 32x16: lane l<16 = column N=l with K 0..7 & 16..23; lanes 16-31 K 8..15 & 24..31.
    // W is [n,k] row-major, so a B column is 32 contiguous ints in LDS.
    v16bf bfrag[2];
#pragma unroll
    for (int j = 0; j < 2; ++j) {
      const int* br = &lB[cur][(waveN + j * 16 + l16) * BK + half * 8];
      put4i(bfrag[j], 0,  *(const int4*)(br + 0));
      put4i(bfrag[j], 4,  *(const int4*)(br + 4));
      put4i(bfrag[j], 8,  *(const int4*)(br + 16));
      put4i(bfrag[j], 12, *(const int4*)(br + 20));
    }

    // ---- 8 WMMAs: 64x32 wave tile, K += 32 ----
#pragma unroll
    for (int i = 0; i < 4; ++i)
#pragma unroll
      for (int j = 0; j < 2; ++j)
        acc[i][j] = __builtin_amdgcn_wmma_f32_16x16x32_bf16(
            /*neg_a=*/false, afrag[i], /*neg_b=*/false, bfrag[j],
            /*c_mod=*/(short)0, acc[i][j], /*reuse_a=*/false, /*reuse_b=*/false);

    __syncthreads();              // all reads of buffer `cur` done before reuse
  }

  // ---- Epilogue: y = scale*acc + bias (f32) ----
  // C/D layout: VGPR r -> M = r + (lane<16 ? 0 : 8), N = lane%16.
  const float s = *scale_p;
#pragma unroll
  for (int i = 0; i < 4; ++i) {
    const int mb = m0 + waveM + i * 16 + half * 8;
#pragma unroll
    for (int j = 0; j < 2; ++j) {
      const int n  = n0 + waveN + j * 16 + l16;
      const float bv = bias[n];
#pragma unroll
      for (int r = 0; r < 8; ++r)
        out[(size_t)(mb + r) * OUT_F + n] = s * acc[i][j][r] + bv;
    }
  }
}

extern "C" void kernel_launch(void* const* d_in, const int* in_sizes, int n_in,
                              void* d_out, int out_size, void* d_ws, size_t ws_size,
                              hipStream_t stream) {
  const float* x    = (const float*)d_in[0];
  const int*   w    = (const int*)d_in[1];   // int8-range values stored as int32
  const float* scal = (const float*)d_in[2]; // 1-element tensor-wise scale
  const float* bias = (const float*)d_in[3];
  float* out = (float*)d_out;

  dim3 grid(OUT_F / BN, MTOT / BM);  // 128 x 64 workgroups
  qlinear_bf16_wmma<<<grid, dim3(256), 0, stream>>>(x, w, scal, bias, out);

  (void)in_sizes; (void)n_in; (void)out_size; (void)d_ws; (void)ws_size;
}
